// EncoderDecoderLSTM_75488345195356
// MI455X (gfx1250) — matrix operand order repaired and checked by reading, not compiled, not run on hardware
//
#include <hip/hip_runtime.h>
#include <hip/hip_bf16.h>

typedef __bf16 bf16_t;
typedef __attribute__((ext_vector_type(8)))  bf16_t bf16x8;
typedef __attribute__((ext_vector_type(16))) bf16_t bf16x16;
typedef __attribute__((ext_vector_type(8)))  float  f32x8;

#define B_      128
#define T_PAST  1024
#define T_TOT   1536     // 1024 past + 512 future
#define IN_F    64
#define IP_     128
#define H_      256
#define G4H     1024     // 4*H
#define KCAT    320      // IN_F + H
#define OUT_F   64
#define AT_PITCH 328     // padded LDS pitch (bf16 elems) for bank-conflict-free b128

// workspace layout (bytes)
#define WS_WCATP 0
#define WS_WCATF (WS_WCATP + G4H*KCAT*2)
#define WS_BIASP (WS_WCATF + G4H*KCAT*2)
#define WS_BIASF (WS_BIASP + G4H*4)
#define WS_DECW  (WS_BIASF + G4H*4)

static __device__ __forceinline__ bf16_t f2bf(float f) {
    union { float f; unsigned u; } v; v.f = f;
    unsigned r = (v.u + 0x7FFFu + ((v.u >> 16) & 1u)) >> 16;   // RNE
    union { unsigned short s; bf16_t b; } o; o.s = (unsigned short)r;
    return o.b;
}

static __device__ __forceinline__ float sigmoidf_(float v) {
    return 1.0f / (1.0f + __expf(-v));
}

static __device__ __forceinline__ bf16x16 cat8(bf16x8 a, bf16x8 b) {
    return __builtin_shufflevector(a, b, 0,1,2,3,4,5,6,7,8,9,10,11,12,13,14,15);
}

// ---------------------------------------------------------------------------
// Phase 1: fold linear layers into concatenated bf16 weight matrices.
//   Wcat_p = [ pW_ih@enc_W | pW_hh ]            (1024 x 320)
//   Wcat_f = [ 0           | fW_ih@fut_W+fW_hh ](1024 x 320)
//   bias_p = pb_ih + pb_hh + pW_ih@enc_b
//   bias_f = fb_ih + fb_hh + fW_ih@fut_b
// ---------------------------------------------------------------------------
__global__ __launch_bounds__(256) void lstm_prep_kernel(
    const float* __restrict__ enc_W, const float* __restrict__ enc_b,
    const float* __restrict__ fut_W, const float* __restrict__ fut_b,
    const float* __restrict__ dec_W,
    const float* __restrict__ pW_ih, const float* __restrict__ pW_hh,
    const float* __restrict__ pb_ih, const float* __restrict__ pb_hh,
    const float* __restrict__ fW_ih, const float* __restrict__ fW_hh,
    const float* __restrict__ fb_ih, const float* __restrict__ fb_hh,
    bf16_t* __restrict__ Wcat_p, bf16_t* __restrict__ Wcat_f,
    float* __restrict__ bias_p, float* __restrict__ bias_f,
    bf16_t* __restrict__ decW)
{
    int idx = blockIdx.x * 256 + threadIdx.x;
    if (idx < G4H * KCAT) {
        int n = idx / KCAT, k = idx % KCAT;
        float vp, vf;
        if (k < IN_F) {
            float s = 0.f;
            for (int j = 0; j < IP_; ++j) s += pW_ih[n*IP_ + j] * enc_W[j*IN_F + k];
            vp = s; vf = 0.f;
        } else {
            int kk = k - IN_F;
            vp = pW_hh[n*H_ + kk];
            float s = fW_hh[n*H_ + kk];
            for (int j = 0; j < IP_; ++j) s += fW_ih[n*IP_ + j] * fut_W[j*H_ + kk];
            vf = s;
        }
        Wcat_p[idx] = f2bf(vp);
        Wcat_f[idx] = f2bf(vf);
    }
    if (idx < G4H) {
        int n = idx;
        float sp = pb_ih[n] + pb_hh[n];
        float sf = fb_ih[n] + fb_hh[n];
        for (int j = 0; j < IP_; ++j) {
            sp += pW_ih[n*IP_ + j] * enc_b[j];
            sf += fW_ih[n*IP_ + j] * fut_b[j];
        }
        bias_p[n] = sp; bias_f[n] = sf;
    }
    if (idx < OUT_F * H_) decW[idx] = f2bf(dec_W[idx]);
}

// ---------------------------------------------------------------------------
// Phase 2: persistent scan. 8 workgroups x 16 batch rows, 1536 steps each.
// Per step: gates[16,1024] = A[16,320] x Wcat^T via v_wmma_f32_16x16x32_bf16,
// in-register LSTM nonlinearity (wave owns same j-slice of all 4 gates),
// fused decoder GEMM writing d_out.
// ---------------------------------------------------------------------------
__global__ __launch_bounds__(256) void lstm_scan_kernel(
    const float* __restrict__ x, const float* __restrict__ dec_b,
    const bf16_t* __restrict__ Wcat_p, const bf16_t* __restrict__ Wcat_f,
    const float* __restrict__ bias_p, const float* __restrict__ bias_f,
    const bf16_t* __restrict__ decW, float* __restrict__ out)
{
    __shared__ bf16_t Atile[16][AT_PITCH];   // [x(64) | h(256)] bf16, padded

    const int tid  = threadIdx.x;
    const int wave = tid >> 5;
    const int lane = tid & 31;
    const int m16  = lane & 15;   // row / column-within-tile per WMMA layout
    const int hsel = lane >> 4;   // half-select
    const int bblk = blockIdx.x * 16;

    // init: zero h region (cols IN_F..AT_PITCH), stage x_0 into cols 0..63
    for (int i = tid; i < 16 * (AT_PITCH - IN_F); i += 256) {
        int m = i / (AT_PITCH - IN_F), c = IN_F + i % (AT_PITCH - IN_F);
        Atile[m][c] = f2bf(0.f);
    }
    for (int i = tid; i < 16 * IN_F; i += 256) {
        int m = i >> 6, k = i & 63;
        Atile[m][k] = f2bf(x[(size_t)(bblk + m) * (T_PAST * IN_F) + k]);
    }

    // per-lane biases: wave w owns gate columns j in [32w, 32w+32)
    float bp[4][2], bfv[4][2];
    #pragma unroll
    for (int g = 0; g < 4; ++g)
        #pragma unroll
        for (int jt = 0; jt < 2; ++jt) {
            int n = g * H_ + wave * 32 + jt * 16 + m16;
            bp[g][jt]  = bias_p[n];
            bfv[g][jt] = bias_f[n];
        }
    float dbias = (wave < 4) ? dec_b[wave * 16 + m16] : 0.f;

    float creg[2][8];   // cell state lives in registers for the whole scan
    #pragma unroll
    for (int jt = 0; jt < 2; ++jt)
        #pragma unroll
        for (int e = 0; e < 8; ++e) creg[jt][e] = 0.f;

    __syncthreads();

    for (int t = 0; t < T_TOT; ++t) {
        const bool past = (t < T_PAST);
        const bf16_t* __restrict__ W = past ? Wcat_p : Wcat_f; // Wcat_f x-cols are 0

        // A fragments (16-bit A 16x32 layout): lane m16, halves depend on hsel
        bf16x16 afrag[10];
        #pragma unroll
        for (int kc = 0; kc < 10; ++kc) {
            int c0 = kc * 32 + hsel * 8;
            afrag[kc] = cat8(*reinterpret_cast<const bf16x8*>(&Atile[m16][c0]),
                             *reinterpret_cast<const bf16x8*>(&Atile[m16][c0 + 16]));
        }

        const int jbase = wave * 32;
        float hval[2][8];

        #pragma unroll
        for (int jt = 0; jt < 2; ++jt) {
            const int ncol = jbase + jt * 16 + m16;

            auto gemmGate = [&](int g) -> f32x8 {
                f32x8 acc = {0.f,0.f,0.f,0.f,0.f,0.f,0.f,0.f};
                const bf16_t* wrow = W + (size_t)(g * H_ + ncol) * KCAT + hsel * 8;
                #pragma unroll
                for (int kc = 0; kc < 10; ++kc) {
                    bf16x16 bfrag = cat8(
                        *reinterpret_cast<const bf16x8*>(wrow + kc * 32),
                        *reinterpret_cast<const bf16x8*>(wrow + kc * 32 + 16));
                    acc = __builtin_amdgcn_wmma_f32_16x16x32_bf16(
                        false, afrag[kc], false, bfrag, (short)0, acc, false, false);
                }
                return acc;
            };

            float si[8], tg[8];
            f32x8 acc = gemmGate(0);                       // i gate
            #pragma unroll
            for (int e = 0; e < 8; ++e)
                si[e] = sigmoidf_(acc[e] + (past ? bp[0][jt] : bfv[0][jt]));
            acc = gemmGate(2);                             // g gate
            #pragma unroll
            for (int e = 0; e < 8; ++e)
                tg[e] = tanhf(acc[e] + (past ? bp[2][jt] : bfv[2][jt]));
            acc = gemmGate(1);                             // f gate -> c update
            #pragma unroll
            for (int e = 0; e < 8; ++e) {
                float c = sigmoidf_(acc[e] + (past ? bp[1][jt] : bfv[1][jt])) * creg[jt][e]
                        + si[e] * tg[e];
                creg[jt][e] = c;
            }
            acc = gemmGate(3);                             // o gate -> h
            #pragma unroll
            for (int e = 0; e < 8; ++e)
                hval[jt][e] = sigmoidf_(acc[e] + (past ? bp[3][jt] : bfv[3][jt]))
                            * tanhf(creg[jt][e]);
        }

        __syncthreads();   // everyone done reading Atile(h_{t-1}, x_t)

        // publish h_t (C/D layout: VGPR e -> row e + 8*hsel, col = owned j + m16)
        #pragma unroll
        for (int jt = 0; jt < 2; ++jt)
            #pragma unroll
            for (int e = 0; e < 8; ++e)
                Atile[e + hsel * 8][IN_F + jbase + jt * 16 + m16] = f2bf(hval[jt][e]);

        // stage x_{t+1}
        if (t + 1 < T_PAST) {
            for (int i = tid; i < 16 * IN_F; i += 256) {
                int m = i >> 6, k = i & 63;
                Atile[m][k] = f2bf(
                    x[(size_t)(bblk + m) * (T_PAST * IN_F) + (size_t)(t + 1) * IN_F + k]);
            }
        }

        __syncthreads();   // Atile now holds h_t (+ x_{t+1})

        // fused decoder: out[b, t, :] = h_t @ dec_W^T + dec_b  (waves 0..3)
        if (wave < 4) {
            f32x8 acc = {0.f,0.f,0.f,0.f,0.f,0.f,0.f,0.f};
            const bf16_t* wrow = decW + (size_t)(wave * 16 + m16) * H_ + hsel * 8;
            #pragma unroll
            for (int kc = 0; kc < 8; ++kc) {
                int c0 = IN_F + kc * 32 + hsel * 8;
                bf16x16 af = cat8(*reinterpret_cast<const bf16x8*>(&Atile[m16][c0]),
                                  *reinterpret_cast<const bf16x8*>(&Atile[m16][c0 + 16]));
                bf16x16 bfrag = cat8(
                    *reinterpret_cast<const bf16x8*>(wrow + kc * 32),
                    *reinterpret_cast<const bf16x8*>(wrow + kc * 32 + 16));
                acc = __builtin_amdgcn_wmma_f32_16x16x32_bf16(
                    false, af, false, bfrag, (short)0, acc, false, false);
            }
            const int ncol = wave * 16 + m16;
            #pragma unroll
            for (int e = 0; e < 8; ++e) {
                int m = e + hsel * 8;
                out[((size_t)(bblk + m) * T_TOT + t) * OUT_F + ncol] = acc[e] + dbias;
            }
        }
    }
}

extern "C" void kernel_launch(void* const* d_in, const int* in_sizes, int n_in,
                              void* d_out, int out_size, void* d_ws, size_t ws_size,
                              hipStream_t stream)
{
    (void)in_sizes; (void)n_in; (void)out_size; (void)ws_size;
    const float* x     = (const float*)d_in[0];
    // d_in[1] = future_n (compile-time constant 512 here)
    const float* enc_W = (const float*)d_in[2];
    const float* enc_b = (const float*)d_in[3];
    const float* fut_W = (const float*)d_in[4];
    const float* fut_b = (const float*)d_in[5];
    const float* dec_W = (const float*)d_in[6];
    const float* dec_b = (const float*)d_in[7];
    const float* pW_ih = (const float*)d_in[8];
    const float* pW_hh = (const float*)d_in[9];
    const float* pb_ih = (const float*)d_in[10];
    const float* pb_hh = (const float*)d_in[11];
    const float* fW_ih = (const float*)d_in[12];
    const float* fW_hh = (const float*)d_in[13];
    const float* fb_ih = (const float*)d_in[14];
    const float* fb_hh = (const float*)d_in[15];

    char* ws = (char*)d_ws;
    bf16_t* Wcat_p = (bf16_t*)(ws + WS_WCATP);
    bf16_t* Wcat_f = (bf16_t*)(ws + WS_WCATF);
    float*  bias_p = (float*)(ws + WS_BIASP);
    float*  bias_f = (float*)(ws + WS_BIASF);
    bf16_t* decW   = (bf16_t*)(ws + WS_DECW);

    const int prepThreads = G4H * KCAT;
    lstm_prep_kernel<<<(prepThreads + 255) / 256, 256, 0, stream>>>(
        enc_W, enc_b, fut_W, fut_b, dec_W, pW_ih, pW_hh, pb_ih, pb_hh,
        fW_ih, fW_hh, fb_ih, fb_hh, Wcat_p, Wcat_f, bias_p, bias_f, decW);

    lstm_scan_kernel<<<8, 256, 0, stream>>>(
        x, dec_b, Wcat_p, Wcat_f, bias_p, bias_f, decW, (float*)d_out);
}
